// QuantumAttentionBlock_50474455663119
// MI455X (gfx1250) — matrix operand order repaired and checked by reading, not compiled, not run on hardware
//
#include <hip/hip_runtime.h>
#include <hip/hip_bf16.h>

// ---------- CDNA5 WMMA types ----------
typedef __attribute__((ext_vector_type(16))) __bf16 v16bf;
typedef __attribute__((ext_vector_type(8)))  float  v8f;
typedef __attribute__((ext_vector_type(4)))  unsigned int u32x4;
typedef __attribute__((ext_vector_type(8)))  int         i32x8;
typedef __attribute__((ext_vector_type(4)))  int         i32x4;

union U16x16 { v16bf v; uint4 q[2]; unsigned int u[8]; unsigned short s[16]; };

#if defined(__has_builtin)
#  if __has_builtin(__builtin_amdgcn_tensor_load_to_lds)
#    define QAB_HAVE_TDM 1
#  else
#    define QAB_HAVE_TDM 0
#  endif
#else
#  define QAB_HAVE_TDM 0
#endif

static __device__ inline v8f wmma_bf16(v16bf a, v16bf b, v8f c) {
    // D = A(16x32 bf16) x B(32x16 bf16) + C(16x16 f32)
    return __builtin_amdgcn_wmma_f32_16x16x32_bf16(
        /*neg_a=*/false, a, /*neg_b=*/false, b,
        /*c_mod=*/(short)0, c, /*reuse_a=*/false, /*reuse_b=*/false);
}

// A-operand (16x32, M x K): lane holds row = lane&15; K halfwords at
// k0 + 8*sub (8 elems) and k0 + 16 + 8*sub (8 elems), sub = lane>>4.
static __device__ inline v16bf load_a(const unsigned short* base, int ld, int k0) {
    int lane = threadIdx.x & 31;
    int r = lane & 15, sub = lane >> 4;
    const unsigned short* p = base + (size_t)r * ld + k0 + 8 * sub;
    U16x16 t;
    t.q[0] = *reinterpret_cast<const uint4*>(p);
    t.q[1] = *reinterpret_cast<const uint4*>(p + 16);
    return t.v;
}

// B-operand (32x16, K x N): lane holds column = lane&15; K halfwords at
// k0 + 8*sub (8 elems) and k0 + 16 + 8*sub (8 elems), sub = lane>>4.
static __device__ inline v16bf load_b(const unsigned short* base, int ld, int k0) {
    int lane = threadIdx.x & 31;
    int r = lane & 15, sub = lane >> 4;
    const unsigned short* p = base + (size_t)r * ld + k0 + 8 * sub;
    U16x16 t;
    t.q[0] = *reinterpret_cast<const uint4*>(p);
    t.q[1] = *reinterpret_cast<const uint4*>(p + 16);
    return t.v;
}

static __device__ inline v16bf neg_bf16x16(v16bf x) {
    U16x16 t; t.v = x;
#pragma unroll
    for (int i = 0; i < 8; i++) t.u[i] ^= 0x80008000u;
    return t.v;
}

static __device__ inline unsigned short f2bf(float f) {
    unsigned int u = __float_as_uint(f);
    return (unsigned short)((u + 0x7FFFu + ((u >> 16) & 1u)) >> 16);
}

static __device__ inline float rmax16(float x) {
#pragma unroll
    for (int m = 8; m >= 1; m >>= 1) x = fmaxf(x, __shfl_xor(x, m, 32));
    return x;
}
static __device__ inline float rsum16(float x) {
#pragma unroll
    for (int m = 8; m >= 1; m >>= 1) x += __shfl_xor(x, m, 32);
    return x;
}

#define V8F_ZERO {0.f,0.f,0.f,0.f,0.f,0.f,0.f,0.f}

// ---------- problem constants ----------
#define BB 2
#define NN 1024
#define CC 1024
#define HH 16
#define DD 64
#define C2 2048   // 2*C
#define ROWS 2048 // B*N

// ---------- TDM: 2-D tile load, descriptor per CDNA5 ISA ----------
// dims/strides in data_size (=2 byte) units; lds_addr/gaddr in bytes.
#if QAB_HAVE_TDM
static __device__ inline void tdm_load_2d(unsigned lds_addr, const void* gaddr,
                                          unsigned tile_d0, unsigned tile_d1,
                                          unsigned tensor_d0, unsigned tensor_d1,
                                          unsigned stride0) {
    unsigned long long ga = (unsigned long long)(size_t)gaddr;
    u32x4 g0;
    g0[0] = 1u;                                   // count=1, user mode, no gather
    g0[1] = lds_addr;                             // LDS byte address
    g0[2] = (unsigned)(ga & 0xffffffffu);         // global_addr[31:0]
    g0[3] = (unsigned)((ga >> 32) & 0x1ffffffu)   // global_addr[56:32]
            | (2u << 30);                         // type=2 ("image")
    i32x8 g1;
    g1[0] = (int)(1u << 16);                      // data_size=1 (2B), wg_mask=0
    g1[1] = (int)((tensor_d0 & 0xffffu) << 16);   // tensor_dim0[15:0]
    g1[2] = (int)((tensor_d0 >> 16) | ((tensor_d1 & 0xffffu) << 16));
    g1[3] = (int)((tensor_d1 >> 16) | (tile_d0 << 16));   // tile_dim0
    g1[4] = (int)(tile_d1 & 0xffffu);             // tile_dim1 (tile_dim2=0)
    g1[5] = (int)stride0;                         // tensor_dim0_stride[31:0]
    g1[6] = 0;                                    // stride0[47:32], stride1[15:0]
    g1[7] = 0;
    i32x4 z4 = {0, 0, 0, 0};
#if __clang_major__ >= 23
    i32x8 z8 = {0, 0, 0, 0, 0, 0, 0, 0};
    __builtin_amdgcn_tensor_load_to_lds(g0, g1, z4, z4, z8, 0);
#else
    __builtin_amdgcn_tensor_load_to_lds(g0, g1, z4, z4, 0);
#endif
}
#endif

// ---------- kernel 1: f32 -> bf16 ----------
__global__ void qab_cvt_bf16(const float* __restrict__ in, unsigned short* __restrict__ out, int n) {
    int i = blockIdx.x * blockDim.x + threadIdx.x;
    if (i < n) out[i] = f2bf(in[i]);
}

// ---------- kernel 2: W (K x N) f32 -> W^T (N x K) bf16 ----------
__global__ void qab_cvt_wT(const float* __restrict__ w, unsigned short* __restrict__ wT, int K, int N) {
    int i = blockIdx.x * blockDim.x + threadIdx.x;
    if (i < K * N) {
        int n = i / K, k = i % K;
        wT[i] = f2bf(w[(size_t)k * N + n]);
    }
}

// ---------- kernel 3: projection GEMM, double-buffered TDM slabs + bf16 WMMA ----------
// Block: 256 threads (8 waves) -> 32x256 output tile. K chunk = 64.
// Per chunk: wave0 issues TDM loads for chunk k+1 into the other LDS buffer,
// waits TENSORcnt<=2 (in-order TDM => chunk k complete), block barriers, all
// waves batch-load 2 A + 8 B operands from LDS and fire 8 WMMAs back-to-back.
// A: ROWS x CC (bf16), Bt: C2 x CC (bf16, row n = column n of W), C: ROWS x C2 (f32)
#define LDS_A_ELEMS (32 * 64)                 // 4 KB
#define LDS_B_ELEMS (256 * 64)                // 32 KB
#define LDS_REGION  (LDS_A_ELEMS + LDS_B_ELEMS) // elements per buffer (36 KB)
__global__ __launch_bounds__(256) void qab_gemm(const unsigned short* __restrict__ A,
                                                const unsigned short* __restrict__ Bt,
                                                float* __restrict__ Cm) {
    extern __shared__ __align__(16) unsigned short smem[];  // 2 x [A slab | B slab]
    int wave = threadIdx.x >> 5;
    int lane = threadIdx.x & 31;
    int mt = wave >> 2;             // 0..1   (16-row subtile)
    int nt = wave & 3;              // 0..3   (64-col subtile)
    int m0 = blockIdx.x * 32;
    int n0 = blockIdx.y * 256;

    v8f acc[4] = {V8F_ZERO, V8F_ZERO, V8F_ZERO, V8F_ZERO};

#if QAB_HAVE_TDM
    unsigned ldsbase = (unsigned)__builtin_amdgcn_groupstaticsize();
    if (threadIdx.x < 32) {         // prologue: chunk 0 -> buffer 0
        tdm_load_2d(ldsbase, A + (size_t)m0 * CC,
                    /*tile*/ 64, 32, /*tensor*/ CC, ROWS, /*stride0*/ CC);
        tdm_load_2d(ldsbase + LDS_A_ELEMS * 2, Bt + (size_t)n0 * CC,
                    /*tile*/ 64, 256, /*tensor*/ CC, C2, /*stride0*/ CC);
    }
#endif
    for (int i = 0; i < CC / 64; i++) {
        int cur = i & 1;
#if QAB_HAVE_TDM
        if (threadIdx.x < 32) {
            if (i < CC / 64 - 1) {  // prefetch chunk i+1 into the other buffer
                unsigned nb = ldsbase + (unsigned)(cur ^ 1) * (LDS_REGION * 2);
                int k1 = (i + 1) * 64;
                tdm_load_2d(nb, A + (size_t)m0 * CC + k1,
                            64, 32, CC, ROWS, CC);
                tdm_load_2d(nb + LDS_A_ELEMS * 2, Bt + (size_t)n0 * CC + k1,
                            64, 256, CC, C2, CC);
                __builtin_amdgcn_s_wait_tensorcnt(2);  // chunk i landed
            } else {
                __builtin_amdgcn_s_wait_tensorcnt(0);
            }
        }
#else
        {
            int k0 = i * 64;
            unsigned short* dstA = smem + cur * LDS_REGION;
            unsigned short* dstB = dstA + LDS_A_ELEMS;
            for (int j = threadIdx.x; j < 32 * 8; j += 256) {
                int r = j >> 3, c = j & 7;
                reinterpret_cast<uint4*>(dstA)[j] =
                    *reinterpret_cast<const uint4*>(A + (size_t)(m0 + r) * CC + k0 + c * 8);
            }
            for (int j = threadIdx.x; j < 256 * 8; j += 256) {
                int r = j >> 3, c = j & 7;
                reinterpret_cast<uint4*>(dstB)[j] =
                    *reinterpret_cast<const uint4*>(Bt + (size_t)(n0 + r) * CC + k0 + c * 8);
            }
        }
#endif
        __syncthreads();            // chunk i visible to all waves
        const unsigned short* As = smem + cur * LDS_REGION + mt * 16 * 64;
        const unsigned short* Bs = smem + cur * LDS_REGION + LDS_A_ELEMS + (nt * 64) * 64;
        // batch all operand loads, then a clean WMMA burst
        v16bf a0 = load_a(As, 64, 0);
        v16bf a1 = load_a(As, 64, 32);
        v16bf b0[4], b1[4];
#pragma unroll
        for (int t = 0; t < 4; t++) {
            b0[t] = load_b(Bs + t * 16 * 64, 64, 0);
            b1[t] = load_b(Bs + t * 16 * 64, 64, 32);
        }
#pragma unroll
        for (int t = 0; t < 4; t++) acc[t] = wmma_bf16(a0, b0[t], acc[t]);
#pragma unroll
        for (int t = 0; t < 4; t++) acc[t] = wmma_bf16(a1, b1[t], acc[t]);
        __syncthreads();            // all waves done with buffer `cur` before reuse
    }
    int col = lane & 15, sub = lane >> 4;
#pragma unroll
    for (int t = 0; t < 4; t++)
#pragma unroll
        for (int r = 0; r < 8; r++)
            Cm[(size_t)(m0 + mt * 16 + r + 8 * sub) * C2 + n0 + nt * 64 + t * 16 + col] = acc[t][r];
}

// ---------- kernel 4: bias + layernorm + relu -> bf16 (natural and/or head-transposed) ----------
__global__ __launch_bounds__(256) void qab_ln_relu(const float* __restrict__ in,
                                                   const float* __restrict__ bias,
                                                   const float* __restrict__ g,
                                                   const float* __restrict__ be,
                                                   unsigned short* outN,   // [row][c], may be null
                                                   unsigned short* outT) { // [b][h][d'][n], may be null
    int row = blockIdx.x, tid = threadIdx.x;
    const float* rp = in + (size_t)row * C2;
    float s = 0.f, s2 = 0.f;
    for (int c = tid; c < C2; c += 256) {
        float t = rp[c] + bias[c];
        s += t; s2 += t * t;
    }
    __shared__ float sh1[256], sh2[256];
    sh1[tid] = s; sh2[tid] = s2;
    __syncthreads();
    for (int o = 128; o > 0; o >>= 1) {
        if (tid < o) { sh1[tid] += sh1[tid + o]; sh2[tid] += sh2[tid + o]; }
        __syncthreads();
    }
    float mean = sh1[0] * (1.f / C2);
    float var  = sh2[0] * (1.f / C2) - mean * mean;
    float inv  = rsqrtf(var + 1e-5f);
    int b = row >> 10, n = row & (NN - 1);
    for (int c = tid; c < C2; c += 256) {
        float t = rp[c] + bias[c];
        float y = fmaxf((t - mean) * inv * g[c] + be[c], 0.f);
        unsigned short bv = f2bf(y);
        if (outN) outN[(size_t)row * C2 + c] = bv;
        if (outT) {
            int h = c >> 7, dp = c & 127;
            outT[(((size_t)(b * HH + h) * 128 + dp) * NN) + n] = bv;
        }
    }
}

// ---------- kernel 5: quantum attention (flash-style, WMMA) ----------
// qb, kb: [B*N][2C] bf16 (per head: d<64 real, d>=64 imag)
// vT:     [B][H][128][N] bf16
// out:    [B][N][2C] f32
__global__ __launch_bounds__(256) void qab_attn(const unsigned short* __restrict__ qb,
                                                const unsigned short* __restrict__ kb,
                                                const unsigned short* __restrict__ vT,
                                                const float* __restrict__ amp_mix,
                                                const float* __restrict__ phase_mix,
                                                float* __restrict__ out) {
    __shared__ __align__(16) unsigned short ldsP[8 * 512]; // 8 waves x (16 rows x 32 cols) bf16
    int wave = threadIdx.x >> 5;
    int lane = threadIdx.x & 31;
    int col = lane & 15, sub = lane >> 4;
    int gid = blockIdx.x * 8 + wave;          // 2048 wave-jobs total
    int bh = gid >> 6, nt = gid & 63;
    int b = bh >> 4, h = bh & 15;
    int n0 = nt * 16;

    const unsigned short* qbase = qb + (size_t)(b * NN + n0) * C2 + h * 128;
    v16bf qr0 = load_a(qbase, C2, 0);
    v16bf qr1 = load_a(qbase, C2, 32);
    v16bf qi0 = load_a(qbase, C2, 64);
    v16bf qi1 = load_a(qbase, C2, 96);
    float amx = amp_mix[h], pmx = phase_mix[h];

    v8f acc[8] = {V8F_ZERO, V8F_ZERO, V8F_ZERO, V8F_ZERO,
                  V8F_ZERO, V8F_ZERO, V8F_ZERO, V8F_ZERO};
    float mrow[8], lrow[8];
#pragma unroll
    for (int r = 0; r < 8; r++) { mrow[r] = -3.4e38f; lrow[r] = 0.f; }

    unsigned short* myP = ldsP + wave * 512;
    v8f z8 = V8F_ZERO;

    for (int m0 = 0; m0 < NN; m0 += 32) {
        float sc[2][8];
#pragma unroll
        for (int t = 0; t < 2; t++) {
            const unsigned short* kbase = kb + (size_t)(b * NN + m0 + t * 16) * C2 + h * 128;
            v16bf kr0 = load_b(kbase, C2, 0);
            v16bf kr1 = load_b(kbase, C2, 32);
            v16bf ki0 = load_b(kbase, C2, 64);
            v16bf ki1 = load_b(kbase, C2, 96);
            // amp = qr.kr - qi.ki  (subtract via sign-flipped bf16 B tile; NEG bits are C-only)
            v8f amp = z8;
            amp = wmma_bf16(qr0, kr0, amp);
            amp = wmma_bf16(qr1, kr1, amp);
            amp = wmma_bf16(qi0, neg_bf16x16(ki0), amp);
            amp = wmma_bf16(qi1, neg_bf16x16(ki1), amp);
            // ph = qr.ki + qi.kr
            v8f ph = z8;
            ph = wmma_bf16(qr0, ki0, ph);
            ph = wmma_bf16(qr1, ki1, ph);
            ph = wmma_bf16(qi0, kr0, ph);
            ph = wmma_bf16(qi1, kr1, ph);
#pragma unroll
            for (int r = 0; r < 8; r++) {
                float a = amp[r] * amx;
                float p = ph[r] * pmx;
                float zv = a * __sinf(p);
                sc[t][r] = a * __cosf(p) * (1.f / (1.f + __expf(-zv)));
            }
        }
        // online softmax across this 32-wide m block (rows: VGPR r -> row r+8*sub)
#pragma unroll
        for (int r = 0; r < 8; r++) {
            float rm = rmax16(fmaxf(sc[0][r], sc[1][r]));
            float mn = fmaxf(mrow[r], rm);
            float scl = __expf(mrow[r] - mn);
            float p0 = __expf(sc[0][r] - mn);
            float p1 = __expf(sc[1][r] - mn);
            lrow[r] = lrow[r] * scl + rsum16(p0 + p1);
            mrow[r] = mn;
#pragma unroll
            for (int t = 0; t < 8; t++) acc[t][r] *= scl;
            int prow = r + 8 * sub;
            myP[prow * 32 + col] = f2bf(p0);
            myP[prow * 32 + 16 + col] = f2bf(p1);
        }
        __builtin_amdgcn_wave_barrier();      // same-wave DS ops are in-order; pin compiler order
        v16bf pa = load_a(myP, 32, 0);        // P as A-operand (16 x 32)
#pragma unroll
        for (int t = 0; t < 8; t++) {
            const unsigned short* vbase = vT + (((size_t)(b * HH + h) * 128 + t * 16) * NN) + m0;
            v16bf bv = load_b(vbase, NN, 0);
            acc[t] = wmma_bf16(pa, bv, acc[t]);
        }
        __builtin_amdgcn_wave_barrier();
    }
#pragma unroll
    for (int t = 0; t < 8; t++) {
#pragma unroll
        for (int r = 0; r < 8; r++) {
            int rr = n0 + r + 8 * sub;
            out[(size_t)(b * NN + rr) * C2 + h * 128 + t * 16 + col] = acc[t][r] / lrow[r];
        }
    }
}

// ---------- host ----------
extern "C" void kernel_launch(void* const* d_in, const int* in_sizes, int n_in,
                              void* d_out, int out_size, void* d_ws, size_t ws_size,
                              hipStream_t stream) {
    const float* x      = (const float*)d_in[0];
    const float* w[3]   = {(const float*)d_in[1], (const float*)d_in[5], (const float*)d_in[9]};
    const float* bia[3] = {(const float*)d_in[2], (const float*)d_in[6], (const float*)d_in[10]};
    const float* gam[3] = {(const float*)d_in[3], (const float*)d_in[7], (const float*)d_in[11]};
    const float* bet[3] = {(const float*)d_in[4], (const float*)d_in[8], (const float*)d_in[12]};
    const float* amp_mix   = (const float*)d_in[13];
    const float* phase_mix = (const float*)d_in[14];
    float* out = (float*)d_out;

    char* ws = (char*)d_ws;
    const size_t MB = 1u << 20;
    unsigned short* xb    = (unsigned short*)(ws + 0);        // 4 MB  (2048x1024 bf16)
    unsigned short* wT[3] = {(unsigned short*)(ws + 4 * MB),  // 3x4MB (2048x1024 bf16 each)
                             (unsigned short*)(ws + 8 * MB),
                             (unsigned short*)(ws + 12 * MB)};
    unsigned short* qbuf  = (unsigned short*)(ws + 16 * MB);  // 8 MB  (2048x2048 bf16)
    unsigned short* kbuf  = (unsigned short*)(ws + 24 * MB);  // 8 MB
    unsigned short* vTbuf = (unsigned short*)(ws + 32 * MB);  // 8 MB  ([2][16][128][1024] bf16)
    float*          scr   = (float*)(ws + 40 * MB);           // 16 MB (2048x2048 f32)

    const int NEL = ROWS * CC; // 2048*1024
    qab_cvt_bf16<<<(NEL + 255) / 256, 256, 0, stream>>>(x, xb, NEL);
    for (int i = 0; i < 3; i++)
        qab_cvt_wT<<<(C2 * CC + 255) / 256, 256, 0, stream>>>(w[i], wT[i], CC, C2);

    // q, k, v projections (scratch reused sequentially)
    unsigned short* natOut[3] = {qbuf, kbuf, nullptr};
    unsigned short* trOut[3]  = {nullptr, nullptr, vTbuf};
    const size_t gemm_lds = 2 * LDS_REGION * sizeof(unsigned short); // 72 KB (double-buffered)
    for (int i = 0; i < 3; i++) {
        qab_gemm<<<dim3(ROWS / 32, C2 / 256), 256, gemm_lds, stream>>>(xb, wT[i], scr);
        qab_ln_relu<<<ROWS, 256, 0, stream>>>(scr, bia[i], gam[i], bet[i], natOut[i], trOut[i]);
    }

    // attention: B*H*(N/16) = 2048 wave-jobs, 8 waves per 256-thread block
    qab_attn<<<256, 256, 0, stream>>>(qbuf, kbuf, vTbuf, amp_mix, phase_mix, out);
    (void)in_sizes; (void)n_in; (void)out_size; (void)ws_size;
}
